// ConditionedInvariantPointAttention_59940563583350
// MI455X (gfx1250) — matrix operand order repaired and checked by reading, not compiled, not run on hardware
//
#include <hip/hip_runtime.h>
#include <math.h>

// ---------------------------------------------------------------------------
// ConditionedInvariantPointAttention on MI455X (gfx1250, wave32, WMMA).
// Bandwidth-bound on z (512*512*128 f32 = 134MB): read it exactly once and
// fuse LN(z) + (z_n @ Wb.T) + (z_n @ Wdz.T). All GEMMs use
// v_wmma_f32_16x16x32_bf16 (bf16 in, f32 accumulate).
// All fragment loaders are branchless; partial tiles use clamp+mask instead
// of exec-divergent guarded loads.
// Workspace requirement: ~17.1M floats (~68.3 MB).
// ---------------------------------------------------------------------------

typedef __attribute__((ext_vector_type(16))) __bf16 v16bf;
typedef __attribute__((ext_vector_type(8)))  float  v8f;

#define NSEQ   512
#define C_S    384
#define C_Z    128
#define C_H    16
#define NH     12
#define P_QK   4
#define P_V    8
#define CZ4    32
#define HC     (NH*C_H)          // 192
#define NN     (NSEQ*NSEQ)       // 262144
#define CATW   960               // 192 + 3*96 + 96 + 384
#define INFV   100000.0f

// ---------------------------- WMMA helpers ---------------------------------

__device__ __forceinline__ v8f wmma32(v16bf a, v16bf b, v8f c) {
  // D = A(16x32) * B(32x16) + C, bf16 inputs, f32 accumulators.
  return __builtin_amdgcn_wmma_f32_16x16x32_bf16(
      /*neg_a=*/false, a, /*neg_b=*/false, b,
      /*c_mod=*/(short)0, c, /*reuse_a=*/false, /*reuse_b=*/false);
}

// A fragment, row-major f32, fully in-bounds 16x32 tile.
// Layout: lane m = lane&15, half=lane>>4; VGPR v holds
// K = half*16 + 2*v and +1 (matches B fragments).
__device__ __forceinline__ v16bf afrag_g(const float* A, long lda, int m0, int k0) {
  int lane = threadIdx.x & 31;
  int m = lane & 15, half = lane >> 4;
  const float* base = A + (long)(m0 + m) * lda + k0;
  v16bf f;
#pragma unroll
  for (int v = 0; v < 8; ++v) {
    int kb = half * 16 + 2 * v;
    f[2*v]   = (__bf16)base[kb];
    f[2*v+1] = (__bf16)base[kb + 1];
  }
  return f;
}

// A fragment with partial rows: branchless clamp + mask (no exec divergence).
__device__ __forceinline__ v16bf afrag_gm(const float* A, long lda, int m0,
                                          int k0, int mMax) {
  int lane = threadIdx.x & 31;
  int m = lane & 15, half = lane >> 4;
  float msk = (m < mMax) ? 1.f : 0.f;
  int mc = (m < mMax) ? m : (mMax - 1);
  const float* base = A + (long)(m0 + mc) * lda + k0;
  v16bf f;
#pragma unroll
  for (int v = 0; v < 8; ++v) {
    int kb = half * 16 + 2 * v;
    f[2*v]   = (__bf16)(base[kb]     * msk);
    f[2*v+1] = (__bf16)(base[kb + 1] * msk);
  }
  return f;
}

// A fragment from bf16 LDS tile (16 x >=k0+32, stride ld elements).
__device__ __forceinline__ v16bf afrag_lds(const __bf16* T, int ld, int k0) {
  int lane = threadIdx.x & 31;
  int m = lane & 15, half = lane >> 4;
  const __bf16* base = T + m * ld + k0;
  v16bf f;
#pragma unroll
  for (int v = 0; v < 8; ++v) {
    int kb = half * 16 + 2 * v;
    f[2*v]   = base[kb];
    f[2*v+1] = base[kb + 1];
  }
  return f;
}

// B = W^T, W row-major [n x k], fully in-bounds tile.
// B 32x16 layout (same k order as A frags): lane n = lane&15; VGPR v holds K = (lane>>4)*16 + 2v,+1.
__device__ __forceinline__ v16bf bfragT(const float* W, long ldw, int n0, int k0) {
  int lane = threadIdx.x & 31;
  int n = lane & 15, half = lane >> 4;
  const float* base = W + (long)(n0 + n) * ldw + k0 + half * 16;
  v16bf f;
#pragma unroll
  for (int v = 0; v < 8; ++v) {
    f[2*v]   = (__bf16)base[2*v];
    f[2*v+1] = (__bf16)base[2*v + 1];
  }
  return f;
}

// B = W^T with partial N: branchless clamp + mask.
__device__ __forceinline__ v16bf bfragT_n(const float* W, long ldw, int n0,
                                          int k0, int nTot) {
  int lane = threadIdx.x & 31;
  int n = lane & 15, half = lane >> 4;
  int row = n0 + n;
  float msk = (row < nTot) ? 1.f : 0.f;
  row = (row < nTot) ? row : (nTot - 1);
  const float* base = W + (long)row * ldw + k0 + half * 16;
  v16bf f;
#pragma unroll
  for (int v = 0; v < 8; ++v) {
    f[2*v]   = (__bf16)(base[2*v]     * msk);
    f[2*v+1] = (__bf16)(base[2*v + 1] * msk);
  }
  return f;
}

// B from row-major f32 B [k x n], fully in-bounds tile.
__device__ __forceinline__ v16bf bfragN(const float* Bm, long ldb, int n0, int k0) {
  int lane = threadIdx.x & 31;
  int n = lane & 15, half = lane >> 4;
  const float* base = Bm + n0 + n + (long)(k0 + half * 16) * ldb;
  v16bf f;
#pragma unroll
  for (int v = 0; v < 8; ++v) {
    f[2*v]   = (__bf16)base[(long)(2*v)     * ldb];
    f[2*v+1] = (__bf16)base[(long)(2*v + 1) * ldb];
  }
  return f;
}

// B row-major with partial N: branchless clamp + mask.
__device__ __forceinline__ v16bf bfragN_n(const float* Bm, long ldb, int n0,
                                          int k0, int nTot) {
  int lane = threadIdx.x & 31;
  int n = lane & 15, half = lane >> 4;
  int col = n0 + n;
  float msk = (col < nTot) ? 1.f : 0.f;
  col = (col < nTot) ? col : (nTot - 1);
  const float* base = Bm + col + (long)(k0 + half * 16) * ldb;
  v16bf f;
#pragma unroll
  for (int v = 0; v < 8; ++v) {
    f[2*v]   = (__bf16)(base[(long)(2*v)     * ldb] * msk);
    f[2*v+1] = (__bf16)(base[(long)(2*v + 1) * ldb] * msk);
  }
  return f;
}

__device__ __forceinline__ float softplusf(float x) {
  return (x > 20.f) ? x : log1pf(__expf(x));
}

// ---------------------- K1: LN(s), LN(cond)*w ------------------------------

__global__ void ln_sc_kernel(const float* __restrict__ s,
                             const float* __restrict__ cond,
                             const float* __restrict__ lnw,
                             float* __restrict__ s_n, float* __restrict__ c_n) {
  int n = blockIdx.x;
  __shared__ float red[128];
  int tid = threadIdx.x;
  auto reduceAll = [&](float v) -> float {
    red[tid] = v; __syncthreads();
    for (int off = 64; off > 0; off >>= 1) {
      if (tid < off) red[tid] += red[tid + off];
      __syncthreads();
    }
    float r = red[0]; __syncthreads();
    return r;
  };
  float a = 0.f, b = 0.f;
  for (int c = tid; c < C_S; c += 128) { float x = s[n * C_S + c]; a += x; b += x * x; }
  float S = reduceAll(a), S2 = reduceAll(b);
  float mu = S / (float)C_S;
  float inv = rsqrtf(S2 / (float)C_S - mu * mu + 1e-5f);
  for (int c = tid; c < C_S; c += 128) s_n[n * C_S + c] = (s[n * C_S + c] - mu) * inv;
  a = 0.f; b = 0.f;
  for (int c = tid; c < C_S; c += 128) { float x = cond[n * C_S + c]; a += x; b += x * x; }
  S = reduceAll(a); S2 = reduceAll(b);
  mu = S / (float)C_S;
  inv = rsqrtf(S2 / (float)C_S - mu * mu + 1e-5f);
  for (int c = tid; c < C_S; c += 128)
    c_n[n * C_S + c] = (cond[n * C_S + c] - mu) * inv * lnw[c];
}

// --------------- K2: s2 = sigmoid(c_n@W1^T + b)*s_n + c_n@W2^T -------------

__global__ void s2_kernel(const float* __restrict__ c_n,
                          const float* __restrict__ s_n,
                          const float* __restrict__ W1,
                          const float* __restrict__ bias,
                          const float* __restrict__ W2,
                          float* __restrict__ s2) {
  int nt = blockIdx.x, mt = blockIdx.y;
  v8f c1 = {}, c2 = {};
  for (int k0 = 0; k0 < C_S; k0 += 32) {
    v16bf a  = afrag_g(c_n, C_S, mt * 16, k0);
    v16bf b1 = bfragT(W1, C_S, nt * 16, k0);
    v16bf b2 = bfragT(W2, C_S, nt * 16, k0);
    c1 = wmma32(a, b1, c1);
    c2 = wmma32(a, b2, c2);
  }
  int lane = threadIdx.x & 31;
  int nn = nt * 16 + (lane & 15), half = lane >> 4;
  float bn = bias[nn];
#pragma unroll
  for (int v = 0; v < 8; ++v) {
    int row = mt * 16 + v + 8 * half;
    float g = 1.f / (1.f + __expf(-(c1[v] + bn)));
    s2[row * C_S + nn] = g * s_n[row * C_S + nn] + c2[v];
  }
}

// ---------- K3: projection out = s2 @ W^T (Nout multiple of 16) ------------

__global__ void proj_kernel(const float* __restrict__ A,
                            const float* __restrict__ W,
                            float* __restrict__ out, int Nout) {
  int nt = blockIdx.x, mt = blockIdx.y;
  v8f c = {};
  for (int k0 = 0; k0 < C_S; k0 += 32) {
    v16bf a = afrag_g(A, C_S, mt * 16, k0);
    v16bf b = bfragT(W, C_S, nt * 16, k0);  // all launches: Nout % 16 == 0
    c = wmma32(a, b, c);
  }
  int lane = threadIdx.x & 31;
  int nn = nt * 16 + (lane & 15), half = lane >> 4;
#pragma unroll
  for (int v = 0; v < 8; ++v) {
    int row = mt * 16 + v + 8 * half;
    out[(long)row * Nout + nn] = c[v];
  }
}

// ---- K4: frames (rot*p+trans), augmented features, row/col point biases ---

__global__ void frames_kernel(const float* __restrict__ qbuf,
                              const float* __restrict__ kvbuf,
                              const float* __restrict__ qp_raw,
                              const float* __restrict__ kvp_raw,
                              const float* __restrict__ rot,
                              const float* __restrict__ trans,
                              const float* __restrict__ hwv,
                              float* __restrict__ vpts,
                              float* __restrict__ featq,
                              float* __restrict__ featk,
                              float* __restrict__ rowb,
                              float* __restrict__ colb) {
  int n = blockIdx.x;
  __shared__ float rb[NH], cb[NH];
  if (threadIdx.x < NH) { rb[threadIdx.x] = 0.f; cb[threadIdx.x] = 0.f; }
  __syncthreads();
  float R[9], T[3];
#pragma unroll
  for (int i = 0; i < 9; ++i) R[i] = rot[n * 9 + i];
#pragma unroll
  for (int i = 0; i < 3; ++i) T[i] = trans[n * 3 + i];
  const float hwscale = sqrtf(1.f / 54.f);  // sqrt(1/(3*(P_QK*9/2)))

  // q points: pflat = h*P_QK + p, raw layout [n][dim*48 + pflat]
  for (int idx = threadIdx.x; idx < NH * P_QK; idx += blockDim.x) {
    int h = idx >> 2, p = idx & 3;
    float px = qp_raw[n * 144 + idx];
    float py = qp_raw[n * 144 + 48 + idx];
    float pz = qp_raw[n * 144 + 96 + idx];
    float x = R[0] * px + R[1] * py + R[2] * pz + T[0];
    float y = R[3] * px + R[4] * py + R[5] * pz + T[1];
    float z = R[6] * px + R[7] * py + R[8] * pz + T[2];
    float hw = softplusf(hwv[h]) * hwscale;
    long fb = (long)h * (NSEQ * 32) + (long)n * 32 + 16 + p * 3;
    featq[fb + 0] = hw * x; featq[fb + 1] = hw * y; featq[fb + 2] = hw * z;
    atomicAdd(&rb[h], -0.5f * hw * (x * x + y * y + z * z));
  }
  // kv points: pflat = h*12 + pp, raw layout [n][dim*144 + pflat]
  for (int idx = threadIdx.x; idx < NH * 12; idx += blockDim.x) {
    int h = idx / 12, pp = idx % 12;
    float px = kvp_raw[n * 432 + idx];
    float py = kvp_raw[n * 432 + 144 + idx];
    float pz = kvp_raw[n * 432 + 288 + idx];
    float x = R[0] * px + R[1] * py + R[2] * pz + T[0];
    float y = R[3] * px + R[4] * py + R[5] * pz + T[1];
    float z = R[6] * px + R[7] * py + R[8] * pz + T[2];
    if (pp < P_QK) {
      long fb = (long)h * (NSEQ * 32) + (long)n * 32 + 16 + pp * 3;
      featk[fb + 0] = x; featk[fb + 1] = y; featk[fb + 2] = z;
      float hw = softplusf(hwv[h]) * hwscale;
      atomicAdd(&cb[h], -0.5f * hw * (x * x + y * y + z * z));
    } else {
      int p = pp - P_QK;
      long vb = (long)n * (NH * P_V * 3) + h * (P_V * 3) + p * 3;
      vpts[vb + 0] = x; vpts[vb + 1] = y; vpts[vb + 2] = z;
    }
  }
  // scalar q/k features (q scaled by 1/sqrt(3*C_H))
  const float qs = rsqrtf(48.f);
  for (int idx = threadIdx.x; idx < HC; idx += blockDim.x) {
    int h = idx >> 4, c = idx & 15;
    long fb = (long)h * (NSEQ * 32) + (long)n * 32 + c;
    featq[fb] = qbuf[n * HC + idx] * qs;
    featk[fb] = kvbuf[n * (2 * HC) + h * 32 + c];
  }
  // zero padding lanes 28..31 of each feature row
  for (int idx = threadIdx.x; idx < NH * 4; idx += blockDim.x) {
    int h = idx >> 2, c = 28 + (idx & 3);
    long fb = (long)h * (NSEQ * 32) + (long)n * 32 + c;
    featq[fb] = 0.f; featk[fb] = 0.f;
  }
  __syncthreads();
  if (threadIdx.x < NH) {
    rowb[threadIdx.x * NSEQ + n] = rb[threadIdx.x];
    colb[threadIdx.x * NSEQ + n] = cb[threadIdx.x];
  }
}

// ------- K5: fused LN(z) + b = z_n@Wb^T + pair_z = z_n@Wdz^T (one z pass) --

__global__ void zfused_kernel(const float* __restrict__ z,
                              const float* __restrict__ lw,
                              const float* __restrict__ lb,
                              const float* __restrict__ Wb,
                              const float* __restrict__ Wdz,
                              float* __restrict__ bbuf,
                              float* __restrict__ pairz) {
  __shared__ __bf16 zt[8][16][132];  // stride 132 bf16 to dodge bank conflicts
  int wave = threadIdx.x >> 5, lane = threadIdx.x & 31;
  long rowbase = (long)blockIdx.x * 128 + wave * 16;
  int quarter = lane & 3;     // 4 lanes per row
  int rsub = lane >> 2;       // 0..7
  for (int pass = 0; pass < 2; ++pass) {
    int rl = pass * 8 + rsub;
    const float* zr = z + (rowbase + rl) * C_Z + quarter * 32;
    __builtin_prefetch(zr + 8 * C_Z, 0, 1);  // rows of next pass
    float vals[32];
    float sm = 0.f, s2 = 0.f;
#pragma unroll
    for (int c = 0; c < 32; ++c) { float x = zr[c]; vals[c] = x; sm += x; s2 += x * x; }
    sm += __shfl_xor(sm, 1); sm += __shfl_xor(sm, 2);
    s2 += __shfl_xor(s2, 1); s2 += __shfl_xor(s2, 2);
    float mu = sm * (1.f / 128.f);
    float inv = rsqrtf(s2 * (1.f / 128.f) - mu * mu + 1e-5f);
#pragma unroll
    for (int c = 0; c < 32; ++c) {
      int col = quarter * 32 + c;
      zt[wave][rl][col] = (__bf16)((vals[c] - mu) * inv * lw[col] + lb[col]);
    }
  }
  __syncthreads();
  v8f c0 = {}, c1 = {}, c2 = {};
#pragma unroll
  for (int k0 = 0; k0 < C_Z; k0 += 32) {
    v16bf a  = afrag_lds(&zt[wave][0][0], 132, k0);
    v16bf b0 = bfragT_n(Wb, C_Z, 0, k0, NH);   // only 12 valid rows
    v16bf b1 = bfragT(Wdz, C_Z, 0,  k0);
    v16bf b2 = bfragT(Wdz, C_Z, 16, k0);
    c0 = wmma32(a, b0, c0);
    c1 = wmma32(a, b1, c1);
    c2 = wmma32(a, b2, c2);
  }
  int nn = lane & 15, half = lane >> 4;
#pragma unroll
  for (int v = 0; v < 8; ++v) {
    long r = rowbase + v + 8 * half;
    if (nn < NH) bbuf[(long)nn * NN + r] = c0[v];
    pairz[r * CZ4 + nn] = c1[v];
    pairz[r * CZ4 + 16 + nn] = c2[v];
  }
}

// ---- K6: logits (one WMMA per 16x16 tile via augmented feats) + softmax ---

__global__ void attn_kernel(const float* __restrict__ featq,
                            const float* __restrict__ featk,
                            const float* __restrict__ bbuf,
                            const float* __restrict__ rowb,
                            const float* __restrict__ colb,
                            const float* __restrict__ mask,
                            float* __restrict__ abuf) {
  int h = blockIdx.x, it = blockIdx.y;
  __shared__ float lg[16 * NSEQ];
  __shared__ float red[16 * 16];
  __shared__ float rowstat[16];
  int wave = threadIdx.x >> 5, lane = threadIdx.x & 31;
  const float invs3 = 0.57735026919f;  // sqrt(1/3)

  v16bf a = afrag_g(featq + (long)h * (NSEQ * 32), 32, it * 16, 0);
#pragma unroll
  for (int t = 0; t < 4; ++t) {
    int j0 = (wave * 4 + t) * 16;
    v16bf b = bfragT(featk + (long)h * (NSEQ * 32), 32, j0, 0);
    v8f c = {};
    c = wmma32(a, b, c);
    int nn = lane & 15, half = lane >> 4;
#pragma unroll
    for (int v = 0; v < 8; ++v) {
      int m = v + 8 * half;
      int i = it * 16 + m, j = j0 + nn;
      float val = c[v] + invs3 * bbuf[(long)h * NN + (long)i * NSEQ + j]
                + rowb[h * NSEQ + i] + colb[h * NSEQ + j]
                + INFV * (mask[i] * mask[j] - 1.f);
      lg[m * NSEQ + j] = val;
    }
  }
  __syncthreads();
  int g = threadIdx.x >> 4, t16 = threadIdx.x & 15;
  float mx = -3.0e38f;
  for (int j = t16; j < NSEQ; j += 16) mx = fmaxf(mx, lg[g * NSEQ + j]);
  red[g * 16 + t16] = mx;
  __syncthreads();
  if (t16 == 0) {
    float m2 = red[g * 16];
    for (int q = 1; q < 16; ++q) m2 = fmaxf(m2, red[g * 16 + q]);
    rowstat[g] = m2;
  }
  __syncthreads();
  float rm = rowstat[g];
  float sm = 0.f;
  for (int j = t16; j < NSEQ; j += 16) {
    float e = __expf(lg[g * NSEQ + j] - rm);
    lg[g * NSEQ + j] = e;
    sm += e;
  }
  __syncthreads();
  red[g * 16 + t16] = sm;
  __syncthreads();
  if (t16 == 0) {
    float s2 = 0.f;
    for (int q = 0; q < 16; ++q) s2 += red[g * 16 + q];
    rowstat[g] = s2;
  }
  __syncthreads();
  float inv = 1.f / rowstat[g];
  for (int j = t16; j < NSEQ; j += 16)
    abuf[(long)h * NN + (long)(it * 16 + g) * NSEQ + j] = lg[g * NSEQ + j] * inv;
}

// ------------------------- K7: o = a @ v -----------------------------------

__global__ void o_kernel(const float* __restrict__ abuf,
                         const float* __restrict__ kvbuf,
                         float* __restrict__ cat) {
  int h = blockIdx.x, it = blockIdx.y;
  const float* A = abuf + (long)h * NN;
  const float* Bm = kvbuf + h * 32 + 16;  // v part, ldb = 2*HC
  v8f c = {};
  for (int k0 = 0; k0 < NSEQ; k0 += 32) {
    __builtin_prefetch(A + (long)(it * 16) * NSEQ + k0 + 128, 0, 1);
    v16bf a = afrag_g(A, NSEQ, it * 16, k0);
    v16bf b = bfragN(Bm, 2 * HC, 0, k0);
    c = wmma32(a, b, c);
  }
  int lane = threadIdx.x & 31;
  int nn = lane & 15, half = lane >> 4;
#pragma unroll
  for (int v = 0; v < 8; ++v) {
    int row = it * 16 + v + 8 * half;
    cat[(long)row * CATW + h * C_H + nn] = c[v];
  }
}

// ----------------------- K8: o_pt raw = a @ v_pts --------------------------

__global__ void opt_kernel(const float* __restrict__ abuf,
                           const float* __restrict__ vpts,
                           float* __restrict__ optb) {
  int h = blockIdx.x, it = blockIdx.y;
  const float* A = abuf + (long)h * NN;
  const float* Bm = vpts + h * (P_V * 3);  // ldb = NH*P_V*3 = 288, N=24
  v8f c0 = {}, c1 = {};
  for (int k0 = 0; k0 < NSEQ; k0 += 32) {
    v16bf a = afrag_g(A, NSEQ, it * 16, k0);
    v16bf b0 = bfragN(Bm, 288, 0, k0);
    v16bf b1 = bfragN_n(Bm, 288, 16, k0, 24);  // cols 24..31 invalid
    c0 = wmma32(a, b0, c0);
    c1 = wmma32(a, b1, c1);
  }
  int lane = threadIdx.x & 31;
  int nn = lane & 15, half = lane >> 4;
#pragma unroll
  for (int v = 0; v < 8; ++v) {
    int row = it * 16 + v + 8 * half;
    optb[(long)row * 288 + h * 24 + nn] = c0[v];
    if (nn < 8) optb[(long)row * 288 + h * 24 + 16 + nn] = c1[v];
  }
}

// --------- K9: inverse frame on o_pt, norms, write into cat ----------------

__global__ void finpt_kernel(const float* __restrict__ optb,
                             const float* __restrict__ rot,
                             const float* __restrict__ trans,
                             float* __restrict__ cat) {
  int n = blockIdx.x;
  for (int idx = threadIdx.x; idx < NH * P_V; idx += blockDim.x) {
    int h = idx >> 3, p = idx & 7;
    float x = optb[(long)n * 288 + h * 24 + p * 3 + 0] - trans[n * 3 + 0];
    float y = optb[(long)n * 288 + h * 24 + p * 3 + 1] - trans[n * 3 + 1];
    float z = optb[(long)n * 288 + h * 24 + p * 3 + 2] - trans[n * 3 + 2];
    // out_i = sum_j rot[n][j][i] * t[j]   (R^T)
    float ox = rot[n * 9 + 0] * x + rot[n * 9 + 3] * y + rot[n * 9 + 6] * z;
    float oy = rot[n * 9 + 1] * x + rot[n * 9 + 4] * y + rot[n * 9 + 7] * z;
    float oz = rot[n * 9 + 2] * x + rot[n * 9 + 5] * y + rot[n * 9 + 8] * z;
    float nr = sqrtf(ox * ox + oy * oy + oz * oz + 1e-8f);
    cat[(long)n * CATW + 192 + idx] = ox;
    cat[(long)n * CATW + 288 + idx] = oy;
    cat[(long)n * CATW + 384 + idx] = oz;
    cat[(long)n * CATW + 480 + idx] = nr;
  }
}

// --------- K10: o_pair[i] = a[:,i,:] @ pair_z[i]  (M=12 heads pad 16) ------

__global__ void opair_kernel(const float* __restrict__ abuf,
                             const float* __restrict__ pairz,
                             float* __restrict__ cat) {
  int i = blockIdx.x;
  const float* A = abuf + (long)i * NSEQ;           // lda = NN, rows = heads
  const float* Bm = pairz + (long)i * (NSEQ * CZ4); // [512 x 32] row-major
  v8f c0 = {}, c1 = {};
  for (int k0 = 0; k0 < NSEQ; k0 += 32) {
    __builtin_prefetch(Bm + (long)(k0 + 32) * CZ4, 0, 1);
    v16bf a = afrag_gm(A, NN, 0, k0, NH);  // rows 12..15 invalid
    v16bf b0 = bfragN(Bm, CZ4, 0, k0);
    v16bf b1 = bfragN(Bm, CZ4, 16, k0);
    c0 = wmma32(a, b0, c0);
    c1 = wmma32(a, b1, c1);
  }
  int lane = threadIdx.x & 31;
  int nn = lane & 15, half = lane >> 4;
#pragma unroll
  for (int v = 0; v < 8; ++v) {
    int m = v + 8 * half;  // head
    if (m < NH) {
      cat[(long)i * CATW + 576 + m * CZ4 + nn] = c0[v];
      cat[(long)i * CATW + 576 + m * CZ4 + 16 + nn] = c1[v];
    }
  }
}

// ------------------------- K11: out = cat @ Wout^T -------------------------

__global__ void out_kernel(const float* __restrict__ cat,
                           const float* __restrict__ Wout,
                           float* __restrict__ out) {
  int nt = blockIdx.x, mt = blockIdx.y;
  v8f c = {};
  for (int k0 = 0; k0 < CATW; k0 += 32) {
    v16bf a = afrag_g(cat, CATW, mt * 16, k0);
    v16bf b = bfragT(Wout, CATW, nt * 16, k0);
    c = wmma32(a, b, c);
  }
  int lane = threadIdx.x & 31;
  int nn = nt * 16 + (lane & 15), half = lane >> 4;
#pragma unroll
  for (int v = 0; v < 8; ++v) {
    int row = mt * 16 + v + 8 * half;
    out[(long)row * C_S + nn] = c[v];
  }
}

// ---------------------------------------------------------------------------

extern "C" void kernel_launch(void* const* d_in, const int* in_sizes, int n_in,
                              void* d_out, int out_size, void* d_ws, size_t ws_size,
                              hipStream_t stream) {
  const float* s        = (const float*)d_in[0];
  const float* cond     = (const float*)d_in[1];
  const float* z        = (const float*)d_in[2];
  const float* rot      = (const float*)d_in[3];
  const float* trans    = (const float*)d_in[4];
  const float* mask     = (const float*)d_in[5];
  const float* ln_cond_w   = (const float*)d_in[6];
  const float* lin_cond_w  = (const float*)d_in[7];
  const float* lin_cond_b  = (const float*)d_in[8];
  const float* lin_cond_nb = (const float*)d_in[9];
  const float* ln_z_w   = (const float*)d_in[10];
  const float* ln_z_b   = (const float*)d_in[11];
  const float* Wq       = (const float*)d_in[12];
  const float* Wkv      = (const float*)d_in[13];
  const float* Wq_pts   = (const float*)d_in[14];
  const float* Wkv_pts  = (const float*)d_in[15];
  const float* Wb       = (const float*)d_in[16];
  const float* Wdz      = (const float*)d_in[17];
  const float* hw       = (const float*)d_in[18];
  const float* Wout     = (const float*)d_in[19];
  float* out = (float*)d_out;

  float* ws = (float*)d_ws;
  size_t off = 0;
  float* s_n   = ws + off; off += (size_t)NSEQ * C_S;
  float* c_n   = ws + off; off += (size_t)NSEQ * C_S;
  float* s2    = ws + off; off += (size_t)NSEQ * C_S;
  float* qbuf  = ws + off; off += (size_t)NSEQ * HC;
  float* kvbuf = ws + off; off += (size_t)NSEQ * 2 * HC;
  float* qp    = ws + off; off += (size_t)NSEQ * 144;
  float* kvp   = ws + off; off += (size_t)NSEQ * 432;
  float* vpts  = ws + off; off += (size_t)NSEQ * NH * P_V * 3;
  float* featq = ws + off; off += (size_t)NH * NSEQ * 32;
  float* featk = ws + off; off += (size_t)NH * NSEQ * 32;
  float* rowb  = ws + off; off += (size_t)NH * NSEQ;
  float* colb  = ws + off; off += (size_t)NH * NSEQ;
  float* bbuf  = ws + off; off += (size_t)NH * NN;
  float* pairz = ws + off; off += (size_t)NN * CZ4;
  float* abuf  = ws + off; off += (size_t)NH * NN;
  float* catb  = ws + off; off += (size_t)NSEQ * CATW;
  float* optb  = ws + off; off += (size_t)NSEQ * 288;
  (void)ws_size; (void)in_sizes; (void)n_in; (void)out_size;

  ln_sc_kernel<<<NSEQ, 128, 0, stream>>>(s, cond, ln_cond_w, s_n, c_n);
  s2_kernel<<<dim3(C_S / 16, NSEQ / 16), 32, 0, stream>>>(
      c_n, s_n, lin_cond_w, lin_cond_b, lin_cond_nb, s2);

  proj_kernel<<<dim3(12, NSEQ / 16), 32, 0, stream>>>(s2, Wq, qbuf, HC);
  proj_kernel<<<dim3(24, NSEQ / 16), 32, 0, stream>>>(s2, Wkv, kvbuf, 2 * HC);
  proj_kernel<<<dim3(9,  NSEQ / 16), 32, 0, stream>>>(s2, Wq_pts, qp, 144);
  proj_kernel<<<dim3(27, NSEQ / 16), 32, 0, stream>>>(s2, Wkv_pts, kvp, 432);

  frames_kernel<<<NSEQ, 128, 0, stream>>>(qbuf, kvbuf, qp, kvp, rot, trans, hw,
                                          vpts, featq, featk, rowb, colb);

  zfused_kernel<<<NN / 128, 256, 0, stream>>>(z, ln_z_w, ln_z_b, Wb, Wdz,
                                              bbuf, pairz);

  attn_kernel<<<dim3(NH, NSEQ / 16), 256, 0, stream>>>(featq, featk, bbuf,
                                                       rowb, colb, mask, abuf);

  o_kernel<<<dim3(NH, NSEQ / 16), 32, 0, stream>>>(abuf, kvbuf, catb);
  opt_kernel<<<dim3(NH, NSEQ / 16), 32, 0, stream>>>(abuf, vpts, optb);
  finpt_kernel<<<NSEQ, 96, 0, stream>>>(optb, rot, trans, catb);
  opair_kernel<<<NSEQ, 32, 0, stream>>>(abuf, pairz, catb);

  out_kernel<<<dim3(C_S / 16, NSEQ / 16), 32, 0, stream>>>(catb, Wout, out);
}
